// THGNN_91061896610342
// MI455X (gfx1250) — compile-verified, matrix operand order checked by reading, not run on hardware
//
#include <hip/hip_runtime.h>
#include <math.h>

// ---------------------------------------------------------------------------
// THGNN forward for MI455X (gfx1250, wave32, WMMA).
// Dense GEMMs: v_wmma_f32_16x16x32_f16, double-buffered LDS pipeline,
// templated epilogues, global_prefetch_b8 hints for the L2-resident tiles.
// ---------------------------------------------------------------------------

typedef __attribute__((ext_vector_type(16))) _Float16 v16h;
typedef __attribute__((ext_vector_type(8)))  float    v8f;

#define LNODES 2048
#define PSTEPS 32
#define FIN    6
#define DMODEL 128
#define TOKENS (LNODES * PSTEPS)   // 65536
#define GDIM   256
#define GHEADS 4
#define GHD    64

#define GSTRIDE 40   // halves per LDS tile row (80 B, multiple of 16 B)

struct alignas(16) h8 { _Float16 h[8]; };

__device__ __forceinline__ float gelu_exact(float x) {
    return 0.5f * x * (1.0f + erff(x * 0.70710678118654752f));
}

// Build one 16x32 f16 WMMA operand fragment from an LDS tile stored row-major
// (32 halves per logical row, GSTRIDE stride, 16B-aligned rows).
// Layout per CDNA5 ISA (16-bit A 16x32): lanes 0-15 hold K{0..7,16..23},
// lanes 16-31 hold K{8..15,24..31}; VGPR v packs halves 2v,2v+1.
__device__ __forceinline__ v16h frag_from_lds(const _Float16* rowp, int lane) {
    int off = (lane & 16) ? 8 : 0;
    h8 u0 = *(const h8*)(rowp + off);
    h8 u1 = *(const h8*)(rowp + 16 + off);
    v16h r;
#pragma unroll
    for (int t = 0; t < 8; ++t) { r[t] = u0.h[t]; r[8 + t] = u1.h[t]; }
    return r;
}

// ---------------------------------------------------------------------------
// Generic WMMA GEMM: C[M,N] = post(A[M,K] @ W[K,N] [+ bias]) [+ residual]
// ACT: 0 = none, 1 = exact GELU, 2 = tanh.  M%128==0, N%128==0, K%32==0.
// Block: 256 threads (8 waves); 128x128 tile; double-buffered LDS staging.
// ---------------------------------------------------------------------------
template <int ACT, bool HAS_BIAS, bool HAS_RES>
__global__ __launch_bounds__(256)
void wmma_gemm_kernel(const float* __restrict__ A, const float* __restrict__ W,
                      const float* __restrict__ bias, const float* __restrict__ residual,
                      float* __restrict__ C, int M, int K, int N)
{
    __shared__ _Float16 As[2][128 * GSTRIDE] __attribute__((aligned(16)));
    __shared__ _Float16 Bs[2][128 * GSTRIDE] __attribute__((aligned(16)));  // [n][k]

    const int tid  = threadIdx.x;
    const int lane = tid & 31;
    const int wave = tid >> 5;
    const int mBase = blockIdx.x * 128;
    const int nBase = blockIdx.y * 128;

    const v8f vzero = {0.f,0.f,0.f,0.f,0.f,0.f,0.f,0.f};
    v8f acc[8];
#pragma unroll
    for (int t = 0; t < 8; ++t) acc[t] = vzero;

    // ---- stage first K-tile into buffer 0 ---------------------------------
#pragma unroll
    for (int i = 0; i < 16; ++i) {
        int idx = tid + i * 256;
        int r = idx >> 5, c = idx & 31;
        As[0][r * GSTRIDE + c] = (_Float16)A[(size_t)(mBase + r) * K + c];
    }
#pragma unroll
    for (int i = 0; i < 16; ++i) {
        int idx = tid + i * 256;
        int r = idx >> 7, c = idx & 127;  // r = k, c = n
        Bs[0][c * GSTRIDE + r] = (_Float16)W[(size_t)r * N + nBase + c];
    }
    __syncthreads();

    int buf = 0;
    for (int kk = 0; kk < K; kk += 32) {
        const bool more = (kk + 32) < K;

        // Prefetch the tile after next into L2 (global_prefetch_b8).
        if (kk + 64 < K) {
            __builtin_prefetch(&A[(size_t)(mBase + (tid >> 1)) * K + kk + 64 + (tid & 1) * 16], 0, 1);
            __builtin_prefetch(&W[(size_t)(kk + 64 + (tid >> 3)) * N + nBase + (tid & 7) * 16], 0, 1);
        }

        // Issue next tile's global loads into registers (hidden under WMMA).
        float aReg[16], bReg[16];
        if (more) {
#pragma unroll
            for (int i = 0; i < 16; ++i) {
                int idx = tid + i * 256;
                int r = idx >> 5, c = idx & 31;
                aReg[i] = A[(size_t)(mBase + r) * K + kk + 32 + c];
            }
#pragma unroll
            for (int i = 0; i < 16; ++i) {
                int idx = tid + i * 256;
                int r = idx >> 7, c = idx & 127;
                bReg[i] = W[(size_t)(kk + 32 + r) * N + nBase + c];
            }
        }

        // WMMA on current buffer: one A fragment, 8 column tiles.
        v16h afrag = frag_from_lds(&As[buf][(wave * 16 + (lane & 15)) * GSTRIDE], lane);
#pragma unroll
        for (int t = 0; t < 8; ++t) {
            v16h bfrag = frag_from_lds(&Bs[buf][(t * 16 + (lane & 15)) * GSTRIDE], lane);
            acc[t] = __builtin_amdgcn_wmma_f32_16x16x32_f16(
                false, afrag, false, bfrag, (short)0, acc[t], false, false);
        }

        // Commit next tile to the alternate buffer.
        if (more) {
#pragma unroll
            for (int i = 0; i < 16; ++i) {
                int idx = tid + i * 256;
                int r = idx >> 5, c = idx & 31;
                As[buf ^ 1][r * GSTRIDE + c] = (_Float16)aReg[i];
            }
#pragma unroll
            for (int i = 0; i < 16; ++i) {
                int idx = tid + i * 256;
                int r = idx >> 7, c = idx & 127;
                Bs[buf ^ 1][c * GSTRIDE + r] = (_Float16)bReg[i];
            }
        }
        __syncthreads();
        buf ^= 1;
    }

    // Epilogue (C layout: lanes 0-15 -> M = r, lanes 16-31 -> M = r + 8)
    const int mOff = wave * 16 + ((lane & 16) ? 8 : 0);
#pragma unroll
    for (int t = 0; t < 8; ++t) {
        const int col = nBase + t * 16 + (lane & 15);
        const float b = HAS_BIAS ? bias[col] : 0.f;
#pragma unroll
        for (int r = 0; r < 8; ++r) {
            size_t row = (size_t)(mBase + mOff + r);
            float v = acc[t][r] + b;
            if (ACT == 1) v = gelu_exact(v);
            if (ACT == 2) v = tanhf(v);
            if (HAS_RES)  v += residual[row * N + col];
            C[row * N + col] = v;
        }
    }
}

// ---------------------------------------------------------------------------
// Input projection + sinusoidal positional encoding: h = x @ in_W + in_b + pe
// ---------------------------------------------------------------------------
__global__ __launch_bounds__(128)
void input_proj_kernel(const float* __restrict__ x, const float* __restrict__ inW,
                       const float* __restrict__ inb, float* __restrict__ h)
{
    const int tok = blockIdx.x;       // 0..65535
    const int d   = threadIdx.x;      // 0..127
    const int p   = tok & (PSTEPS - 1);
    float acc = inb[d];
#pragma unroll
    for (int f = 0; f < FIN; ++f) acc += x[(size_t)tok * FIN + f] * inW[f * DMODEL + d];
    const int i2 = d >> 1;
    const float div = __expf(-(float)(2 * i2) * (9.210340371976184f / 128.f));
    const float ang = (float)p * div;
    acc += (d & 1) ? __cosf(ang) : __sinf(ang);
    h[(size_t)tok * DMODEL + d] = acc;
}

// ---------------------------------------------------------------------------
// LayerNorm over D=128; one wave32 per row, 8 rows per 256-thread block.
// ---------------------------------------------------------------------------
__global__ __launch_bounds__(256)
void layernorm_kernel(const float* __restrict__ X, const float* __restrict__ g,
                      const float* __restrict__ b, float* __restrict__ Y, int M)
{
    const int row  = blockIdx.x * 8 + (threadIdx.x >> 5);
    const int lane = threadIdx.x & 31;
    if (row >= M) return;
    const float* xr = X + (size_t)row * DMODEL;
    float4 v = *(const float4*)(xr + lane * 4);
    float s = v.x + v.y + v.z + v.w;
#pragma unroll
    for (int o = 16; o > 0; o >>= 1) s += __shfl_xor(s, o, 32);
    const float mean = s * (1.f / 128.f);
    float d0 = v.x - mean, d1 = v.y - mean, d2 = v.z - mean, d3 = v.w - mean;
    float q = d0 * d0 + d1 * d1 + d2 * d2 + d3 * d3;
#pragma unroll
    for (int o = 16; o > 0; o >>= 1) q += __shfl_xor(q, o, 32);
    const float inv = rsqrtf(q * (1.f / 128.f) + 1e-5f);
    float* yr = Y + (size_t)row * DMODEL;
    const int c = lane * 4;
    yr[c + 0] = d0 * inv * g[c + 0] + b[c + 0];
    yr[c + 1] = d1 * inv * g[c + 1] + b[c + 1];
    yr[c + 2] = d2 * inv * g[c + 2] + b[c + 2];
    yr[c + 3] = d3 * inv * g[c + 3] + b[c + 3];
}

// ---------------------------------------------------------------------------
// Per-node MHA: 8 heads, head dim 16, sequence P=32. One block per node.
// ---------------------------------------------------------------------------
__global__ __launch_bounds__(256)
void mha_attention_kernel(const float* __restrict__ Q, const float* __restrict__ K,
                          const float* __restrict__ V, float* __restrict__ O)
{
    __shared__ float Qs[32][128], Ks[32][128], Vs[32][128];
    __shared__ float Sc[32][33];
    const int l = blockIdx.x;
    const int tid = threadIdx.x;
    const size_t base = (size_t)l * 32 * 128;
    for (int i = tid; i < 4096; i += 256) {
        int r = i >> 7, c = i & 127;
        Qs[r][c] = Q[base + i]; Ks[r][c] = K[base + i]; Vs[r][c] = V[base + i];
    }
    __syncthreads();
    for (int h = 0; h < 8; ++h) {
        const int hb = h * 16;
        for (int i = tid; i < 1024; i += 256) {
            int qi = i >> 5, ki = i & 31;
            float s = 0.f;
#pragma unroll
            for (int d = 0; d < 16; ++d) s += Qs[qi][hb + d] * Ks[ki][hb + d];
            Sc[qi][ki] = s * 0.25f;   // 1/sqrt(16)
        }
        __syncthreads();
        if (tid < 32) {
            float m = -1e30f;
            for (int k2 = 0; k2 < 32; ++k2) m = fmaxf(m, Sc[tid][k2]);
            float z = 0.f;
            for (int k2 = 0; k2 < 32; ++k2) { float e = __expf(Sc[tid][k2] - m); Sc[tid][k2] = e; z += e; }
            const float r = 1.f / z;
            for (int k2 = 0; k2 < 32; ++k2) Sc[tid][k2] *= r;
        }
        __syncthreads();
        for (int i = tid; i < 512; i += 256) {
            int qi = i >> 4, d = i & 15;
            float o = 0.f;
            for (int k2 = 0; k2 < 32; ++k2) o += Sc[qi][k2] * Vs[k2][hb + d];
            O[base + (size_t)qi * 128 + hb + d] = o;
        }
        __syncthreads();
    }
}

__global__ __launch_bounds__(128)
void take_last_kernel(const float* __restrict__ h, float* __restrict__ henc)
{
    const int l = blockIdx.x, d = threadIdx.x;
    henc[(size_t)l * DMODEL + d] = h[((size_t)l * PSTEPS + (PSTEPS - 1)) * DMODEL + d];
}

// ---------------------------------------------------------------------------
// GAT: per-(node,head) attention scores  si = <Wh, a_i>, sj = <Wh, a_j>
// ---------------------------------------------------------------------------
__global__ __launch_bounds__(256)
void gat_scores_kernel(const float* __restrict__ Wh, const float* __restrict__ attn,
                       float* __restrict__ si, float* __restrict__ sj)
{
    const int idx  = blockIdx.x * 8 + (threadIdx.x >> 5);  // l*4 + h
    const int lane = threadIdx.x & 31;
    const int l = idx >> 2, h = idx & 3;
    const float* w  = Wh + (size_t)l * GDIM + h * GHD;
    const float* ai = attn + h * (2 * GHD);
    const float* aj = ai + GHD;
    float s1 = w[lane] * ai[lane] + w[lane + 32] * ai[lane + 32];
    float s2 = w[lane] * aj[lane] + w[lane + 32] * aj[lane + 32];
#pragma unroll
    for (int o = 16; o > 0; o >>= 1) { s1 += __shfl_xor(s1, o, 32); s2 += __shfl_xor(s2, o, 32); }
    if (lane == 0) { si[idx] = s1; sj[idx] = s2; }
}

// Pass 1: masked-softmax statistics (max m, partition Z) per (i, head).
__global__ __launch_bounds__(256)
void gat_pass1_kernel(const int* __restrict__ A, const float* __restrict__ si,
                      const float* __restrict__ sj, float* __restrict__ m, float* __restrict__ Z)
{
    __shared__ float red[256];
    const int i = blockIdx.x, tid = threadIdx.x;
    float s_i[4];
#pragma unroll
    for (int h = 0; h < 4; ++h) s_i[h] = si[i * 4 + h];
    const int* Ar = A + (size_t)i * LNODES;

    float lmax[4] = {-1e30f, -1e30f, -1e30f, -1e30f};
    for (int j = tid; j < LNODES; j += 256) {
        if (Ar[j] > 0) {
#pragma unroll
            for (int h = 0; h < 4; ++h) {
                float e = s_i[h] + sj[j * 4 + h];
                e = e > 0.f ? e : 0.2f * e;       // leaky_relu(0.2)
                lmax[h] = fmaxf(lmax[h], e);
            }
        }
    }
    float mh[4];
#pragma unroll
    for (int h = 0; h < 4; ++h) {
        red[tid] = lmax[h]; __syncthreads();
        for (int o = 128; o > 0; o >>= 1) { if (tid < o) red[tid] = fmaxf(red[tid], red[tid + o]); __syncthreads(); }
        mh[h] = red[0]; __syncthreads();
    }
    float lsum[4] = {0.f, 0.f, 0.f, 0.f};
    for (int j = tid; j < LNODES; j += 256) {
        if (Ar[j] > 0) {
#pragma unroll
            for (int h = 0; h < 4; ++h) {
                float e = s_i[h] + sj[j * 4 + h];
                e = e > 0.f ? e : 0.2f * e;
                lsum[h] += __expf(e - mh[h]);
            }
        }
    }
#pragma unroll
    for (int h = 0; h < 4; ++h) {
        red[tid] = lsum[h]; __syncthreads();
        for (int o = 128; o > 0; o >>= 1) { if (tid < o) red[tid] += red[tid + o]; __syncthreads(); }
        if (tid == 0) { m[i * 4 + h] = mh[h]; Z[i * 4 + h] = red[0]; }
        __syncthreads();
    }
}

// Pass 2: out[i, h*64+d] = sum_j alpha[i,j,h] * Wh[j, h*64+d] via WMMA.
// Grid: (L/16, GHEADS); block 128 threads = 4 waves, one 16-col tile each.
__global__ __launch_bounds__(128)
void gat_pass2_kernel(const int* __restrict__ A, const float* __restrict__ si,
                      const float* __restrict__ sj, const float* __restrict__ m,
                      const float* __restrict__ Z, const float* __restrict__ Wh,
                      float* __restrict__ out)
{
    __shared__ _Float16 wT[16 * GSTRIDE] __attribute__((aligned(16)));
    __shared__ _Float16 BT[64 * GSTRIDE] __attribute__((aligned(16)));
    __shared__ float sm[16], sz[16], ss[16];

    const int iBase = blockIdx.x * 16;
    const int h = blockIdx.y;
    const int tid = threadIdx.x, lane = tid & 31, wave = tid >> 5;

    if (tid < 16) {
        sm[tid] = m[(iBase + tid) * 4 + h];
        float z = Z[(iBase + tid) * 4 + h];
        sz[tid] = (z > 0.f) ? (1.f / z) : 0.f;   // no neighbors -> row of zeros
        ss[tid] = si[(iBase + tid) * 4 + h];
    }
    __syncthreads();

    const v8f vzero = {0.f,0.f,0.f,0.f,0.f,0.f,0.f,0.f};
    v8f acc = vzero;

    for (int jb = 0; jb < LNODES; jb += 32) {
        // alpha tile 16(i) x 32(j) in f16
        for (int idx = tid; idx < 512; idx += 128) {
            int il = idx >> 5, jl = idx & 31;
            int j = jb + jl;
            float wv = 0.f;
            if (A[(size_t)(iBase + il) * LNODES + j] > 0) {
                float e = ss[il] + sj[j * 4 + h];
                e = e > 0.f ? e : 0.2f * e;
                wv = __expf(e - sm[il]) * sz[il];
            }
            wT[il * GSTRIDE + jl] = (_Float16)wv;
        }
        // Wh tile transposed: BT[d][jl], d in [0,64)
        for (int idx = tid; idx < 2048; idx += 128) {
            int jl = idx >> 6, d = idx & 63;
            BT[d * GSTRIDE + jl] = (_Float16)Wh[(size_t)(jb + jl) * GDIM + h * GHD + d];
        }
        __syncthreads();
        v16h af = frag_from_lds(&wT[(lane & 15) * GSTRIDE], lane);
        v16h bf = frag_from_lds(&BT[(wave * 16 + (lane & 15)) * GSTRIDE], lane);
        acc = __builtin_amdgcn_wmma_f32_16x16x32_f16(false, af, false, bf, (short)0, acc, false, false);
        __syncthreads();
    }

    const int mOff = (lane & 16) ? 8 : 0;
    const int col = h * GHD + wave * 16 + (lane & 15);
#pragma unroll
    for (int r = 0; r < 8; ++r)
        out[(size_t)(iBase + mOff + r) * GDIM + col] = acc[r];
}

// ---------------------------------------------------------------------------
// Hetero aggregation tails + classifier head
// ---------------------------------------------------------------------------
__global__ __launch_bounds__(256)
void rowdot_kernel(const float* __restrict__ T, const float* __restrict__ q,
                   float* __restrict__ rd, int M)
{
    const int row = blockIdx.x * 8 + (threadIdx.x >> 5);
    const int lane = threadIdx.x & 31;
    if (row >= M) return;
    const float* tr = T + (size_t)row * 256;
    float s = 0.f;
#pragma unroll
    for (int k = 0; k < 8; ++k) s += tr[lane + k * 32] * q[lane + k * 32];
#pragma unroll
    for (int o = 16; o > 0; o >>= 1) s += __shfl_xor(s, o, 32);
    if (lane == 0) rd[row] = s;
}

__global__ __launch_bounds__(256)
void het_weights_kernel(const float* __restrict__ rd, float* __restrict__ wout)
{
    __shared__ float red[256];
    float acc[3];
    for (int c = 0; c < 3; ++c) {
        float s = 0.f;
        for (int j = threadIdx.x; j < LNODES; j += 256) s += rd[c * LNODES + j];
        red[threadIdx.x] = s; __syncthreads();
        for (int o = 128; o > 0; o >>= 1) { if (threadIdx.x < o) red[threadIdx.x] += red[threadIdx.x + o]; __syncthreads(); }
        acc[c] = red[0] * (1.f / (float)LNODES); __syncthreads();
    }
    if (threadIdx.x == 0) {
        float mx = fmaxf(acc[0], fmaxf(acc[1], acc[2]));
        float e0 = __expf(acc[0] - mx), e1 = __expf(acc[1] - mx), e2 = __expf(acc[2] - mx);
        float inv = 1.f / (e0 + e1 + e2);
        wout[0] = e0 * inv; wout[1] = e1 * inv; wout[2] = e2 * inv;
    }
}

__global__ __launch_bounds__(256)
void het_combine_kernel(const float* __restrict__ w, const float* __restrict__ hs,
                        const float* __restrict__ hp, const float* __restrict__ hn,
                        float* __restrict__ z)
{
    const size_t i = (size_t)blockIdx.x * 256 + threadIdx.x;
    z[i] = w[0] * hs[i] + w[1] * hp[i] + w[2] * hn[i];
}

__global__ __launch_bounds__(256)
void classifier_final_kernel(const float* __restrict__ z1, const float* __restrict__ W2,
                             const float* __restrict__ b2, float* __restrict__ out)
{
    const int row = blockIdx.x * 8 + (threadIdx.x >> 5);
    const int lane = threadIdx.x & 31;
    const float* zr = z1 + (size_t)row * 256;
    float s = 0.f;
#pragma unroll
    for (int k = 0; k < 8; ++k) s += zr[lane + k * 32] * W2[lane + k * 32];
#pragma unroll
    for (int o = 16; o > 0; o >>= 1) s += __shfl_xor(s, o, 32);
    if (lane == 0) out[row] = 1.f / (1.f + __expf(-(s + b2[0])));
}

// ---------------------------------------------------------------------------
// Host-side GEMM dispatcher over template specializations.
// ---------------------------------------------------------------------------
static inline void gemm_dispatch(const float* A, const float* W, const float* b,
                                 const float* res, float* C, int M, int K, int N,
                                 int act, hipStream_t stream)
{
    dim3 grid(M / 128, N / 128);
    if (res)            wmma_gemm_kernel<0, true,  true ><<<grid, 256, 0, stream>>>(A, W, b, res, C, M, K, N);
    else if (!b)        wmma_gemm_kernel<0, false, false><<<grid, 256, 0, stream>>>(A, W, nullptr, nullptr, C, M, K, N);
    else if (act == 1)  wmma_gemm_kernel<1, true,  false><<<grid, 256, 0, stream>>>(A, W, b, nullptr, C, M, K, N);
    else if (act == 2)  wmma_gemm_kernel<2, true,  false><<<grid, 256, 0, stream>>>(A, W, b, nullptr, C, M, K, N);
    else                wmma_gemm_kernel<0, true,  false><<<grid, 256, 0, stream>>>(A, W, b, nullptr, C, M, K, N);
}

// ---------------------------------------------------------------------------
// Host orchestration
// ---------------------------------------------------------------------------
extern "C" void kernel_launch(void* const* d_in, const int* in_sizes, int n_in,
                              void* d_out, int out_size, void* d_ws, size_t ws_size,
                              hipStream_t stream)
{
    (void)in_sizes; (void)n_in; (void)out_size; (void)ws_size;

    // ---- input unpacking (setup_inputs dict order, nested dicts flattened) --
    const float* x      = (const float*)d_in[0];
    const int*   A_pos  = (const int*)  d_in[1];
    const int*   A_neg  = (const int*)  d_in[2];
    const float* in_W   = (const float*)d_in[3];
    const float* in_b   = (const float*)d_in[4];
    const float* LP[2][16];
    for (int l = 0; l < 2; ++l)
        for (int j = 0; j < 16; ++j)
            LP[l][j] = (const float*)d_in[5 + l * 16 + j];
    enum { LN1G=0, LN1B, WQ, BQ, WK, BK, WV, BV, WO, BO, LN2G, LN2B, FW1, FB1, FW2, FB2 };
    const float* self_W = (const float*)d_in[37];
    const float* self_b = (const float*)d_in[38];
    const float* gW[2]    = {(const float*)d_in[39], (const float*)d_in[43]};
    const float* gAttn[2] = {(const float*)d_in[40], (const float*)d_in[44]};
    const float* gOW[2]   = {(const float*)d_in[41], (const float*)d_in[45]};
    const float* gOB[2]   = {(const float*)d_in[42], (const float*)d_in[46]};
    const int*   gAdj[2]  = {A_pos, A_neg};
    const float* Ws = (const float*)d_in[47]; const float* bs = (const float*)d_in[48];
    const float* Wp = (const float*)d_in[49]; const float* bp = (const float*)d_in[50];
    const float* Wn = (const float*)d_in[51]; const float* bn = (const float*)d_in[52];
    const float* qv = (const float*)d_in[53];
    const float* cls_W1 = (const float*)d_in[54]; const float* cls_b1 = (const float*)d_in[55];
    const float* cls_W2 = (const float*)d_in[56]; const float* cls_b2 = (const float*)d_in[57];
    float* out = (float*)d_out;

    // ---- workspace carving (~220 MB total) --------------------------------
    float* ws = (float*)d_ws;
    const size_t T = (size_t)TOKENS;
    float* h    = ws;                // 65536x128
    float* hn   = ws + T * 128;      // 65536x128
    float* Qb   = ws + T * 256;      // 65536x128
    float* Kb   = ws + T * 384;      // 65536x128
    float* Vb   = ws + T * 512;      // 65536x128
    float* ff   = ws + T * 256;      // 65536x512 (reuses Q/K/V region after attention)
    float* p    = ws + T * 768;      // small region (192 MB mark)
    float* h_enc  = p; p += (size_t)LNODES * 128;
    float* h_self = p; p += (size_t)LNODES * 256;
    float* Wh     = p; p += (size_t)LNODES * 256;   // reused per relation
    float* si     = p; p += (size_t)LNODES * 4;
    float* sj     = p; p += (size_t)LNODES * 4;
    float* mM     = p; p += (size_t)LNODES * 4;
    float* Zz     = p; p += (size_t)LNODES * 4;
    float* aggBuf = p; p += (size_t)LNODES * 256;   // reused per relation
    float* gout[2]; gout[0] = p; p += (size_t)LNODES * 256;
                    gout[1] = p; p += (size_t)LNODES * 256;
    float* t_s    = p; p += (size_t)LNODES * 256;
    float* t_p    = p; p += (size_t)LNODES * 256;
    float* t_n    = p; p += (size_t)LNODES * 256;
    float* rd     = p; p += 3 * (size_t)LNODES;
    float* wcoef  = p; p += 4;
    float* zbuf   = p; p += (size_t)LNODES * 256;
    float* z1     = p; p += (size_t)LNODES * 256;

    // ---- 1) input projection + positional encoding ------------------------
    input_proj_kernel<<<TOKENS, 128, 0, stream>>>(x, in_W, in_b, h);

    // ---- 2) two pre-LN transformer encoder layers -------------------------
    for (int l = 0; l < 2; ++l) {
        layernorm_kernel<<<TOKENS / 8, 256, 0, stream>>>(h, LP[l][LN1G], LP[l][LN1B], hn, TOKENS);
        gemm_dispatch(hn, LP[l][WQ], LP[l][BQ], nullptr, Qb, TOKENS, 128, 128, 0, stream);
        gemm_dispatch(hn, LP[l][WK], LP[l][BK], nullptr, Kb, TOKENS, 128, 128, 0, stream);
        gemm_dispatch(hn, LP[l][WV], LP[l][BV], nullptr, Vb, TOKENS, 128, 128, 0, stream);
        mha_attention_kernel<<<LNODES, 256, 0, stream>>>(Qb, Kb, Vb, hn);   // attn out -> hn
        gemm_dispatch(hn, LP[l][WO], LP[l][BO], h, h, TOKENS, 128, 128, 0, stream);   // + residual
        layernorm_kernel<<<TOKENS / 8, 256, 0, stream>>>(h, LP[l][LN2G], LP[l][LN2B], hn, TOKENS);
        gemm_dispatch(hn, LP[l][FW1], LP[l][FB1], nullptr, ff, TOKENS, 128, 512, 1, stream); // GELU
        gemm_dispatch(ff, LP[l][FW2], LP[l][FB2], h, h, TOKENS, 512, 128, 0, stream);        // + residual
    }
    take_last_kernel<<<LNODES, 128, 0, stream>>>(h, h_enc);

    // ---- 3) self projection ------------------------------------------------
    gemm_dispatch(h_enc, self_W, self_b, nullptr, h_self, LNODES, 128, 256, 0, stream);

    // ---- 4) two GAT relations (masked softmax + WMMA aggregation) ----------
    for (int rel = 0; rel < 2; ++rel) {
        gemm_dispatch(h_enc, gW[rel], nullptr, nullptr, Wh, LNODES, 128, 256, 0, stream);
        gat_scores_kernel<<<LNODES * GHEADS / 8, 256, 0, stream>>>(Wh, gAttn[rel], si, sj);
        gat_pass1_kernel<<<LNODES, 256, 0, stream>>>(gAdj[rel], si, sj, mM, Zz);
        gat_pass2_kernel<<<dim3(LNODES / 16, GHEADS), 128, 0, stream>>>(
            gAdj[rel], si, sj, mM, Zz, Wh, aggBuf);
        gemm_dispatch(aggBuf, gOW[rel], gOB[rel], nullptr, gout[rel], LNODES, 256, 256, 0, stream);
    }

    // ---- 5) hetero attention aggregation -----------------------------------
    gemm_dispatch(h_self,  Ws, bs, nullptr, t_s, LNODES, 256, 256, 2, stream);   // tanh
    gemm_dispatch(gout[0], Wp, bp, nullptr, t_p, LNODES, 256, 256, 2, stream);
    gemm_dispatch(gout[1], Wn, bn, nullptr, t_n, LNODES, 256, 256, 2, stream);
    rowdot_kernel<<<LNODES / 8, 256, 0, stream>>>(t_s, qv, rd + 0 * LNODES, LNODES);
    rowdot_kernel<<<LNODES / 8, 256, 0, stream>>>(t_p, qv, rd + 1 * LNODES, LNODES);
    rowdot_kernel<<<LNODES / 8, 256, 0, stream>>>(t_n, qv, rd + 2 * LNODES, LNODES);
    het_weights_kernel<<<1, 256, 0, stream>>>(rd, wcoef);
    het_combine_kernel<<<LNODES, 256, 0, stream>>>(wcoef, h_self, gout[0], gout[1], zbuf);

    // ---- 6) classifier head ------------------------------------------------
    gemm_dispatch(zbuf, cls_W1, cls_b1, nullptr, z1, LNODES, 256, 256, 1, stream);  // GELU
    classifier_final_kernel<<<LNODES / 8, 256, 0, stream>>>(z1, cls_W2, cls_b2, out);
}
